// SoftHebbLayer_7172595385011
// MI455X (gfx1250) — matrix operand order repaired and checked by reading, not compiled, not run on hardware
//
#include <hip/hip_runtime.h>
#include <hip/hip_bf16.h>

#define EPS 1e-9f

typedef __attribute__((ext_vector_type(16))) __bf16 v16bf;
typedef __attribute__((ext_vector_type(8)))  __bf16 v8bf;
typedef __attribute__((ext_vector_type(8)))  float  v8f;

// ---------------------------------------------------------------- reductions
__device__ __forceinline__ float block_reduce_sum256(float v) {
    __shared__ float s[8];
    #pragma unroll
    for (int o = 16; o > 0; o >>= 1) v += __shfl_xor(v, o, 32);
    const int lane = threadIdx.x & 31;
    const int wid  = threadIdx.x >> 5;
    if (lane == 0) s[wid] = v;
    __syncthreads();
    if (wid == 0) {
        v = (lane < 8) ? s[lane] : 0.0f;
        #pragma unroll
        for (int o = 4; o > 0; o >>= 1) v += __shfl_xor(v, o, 32);
        if (lane == 0) s[0] = v;
    }
    __syncthreads();
    return s[0];
}

// ---------------------------------------------------------------- init
__global__ void init_max_kernel(unsigned* __restrict__ m) { *m = 0u; }

// ------------------------------------------------- row L2 normalize -> bf16
__global__ void __launch_bounds__(256)
row_l2_norm_bf16_kernel(const float* __restrict__ src, __bf16* __restrict__ dst, int ncols) {
    const size_t row = blockIdx.x;
    const float* s = src + row * (size_t)ncols;
    __bf16*      d = dst + row * (size_t)ncols;
    float ss = 0.0f;
    for (int c = threadIdx.x; c < ncols; c += blockDim.x) {
        float v = s[c];
        ss = fmaf(v, v, ss);
    }
    ss = block_reduce_sum256(ss);
    const float inv = 1.0f / (sqrtf(ss) + EPS);
    for (int c = threadIdx.x; c < ncols; c += blockDim.x)
        d[c] = (__bf16)(s[c] * inv);
}

// ---------------------------------------------------------------- WMMA GEMM
// A fragment (16x32 bf16, M x K): lane l holds row m0+(l&15); with h = l>>4 the
// lane's 16 values are K = k0 + h*16 + {0..15}.
__device__ __forceinline__ v16bf load_a_tile(const __bf16* p) {
    union { v16bf v; v8bf h[2]; } u;
    u.h[0] = *(const v8bf*)(p);
    u.h[1] = *(const v8bf*)(p + 8);
    return u.v;
}

// Wave tile: 64(M) x 64(N) = 4x4 WMMA fragments, 16 wmma per 16 b128 loads.
// Block: 8 waves in 2(M) x 4(N) -> 128 x 256 block tile.
__global__ void __launch_bounds__(256)
softhebb_wmma_gemm(const __bf16* __restrict__ X,   // [M,K] row-major, row-normalized
                   const __bf16* __restrict__ W,   // [N,K] row-major, row-normalized
                   float* __restrict__ out,        // [M,N] = relu(X @ W^T)
                   int K, int N,
                   unsigned* __restrict__ maxbits) {
    const int lane = threadIdx.x & 31;
    const int wv   = threadIdx.x >> 5;        // wave 0..7, arranged 2(M) x 4(N)
    const int r    = lane & 15;
    const int h    = lane >> 4;
    const int m0   = blockIdx.x * 128 + (wv & 1)  * 64;
    const int n0   = blockIdx.y * 256 + (wv >> 1) * 64;

    const v8f zero = {};
    v8f acc[4][4];
    #pragma unroll
    for (int i = 0; i < 4; ++i)
        #pragma unroll
        for (int j = 0; j < 4; ++j) acc[i][j] = zero;

    const __bf16* xa[4];
    const __bf16* wb[4];
    #pragma unroll
    for (int i = 0; i < 4; ++i)
        xa[i] = X + (size_t)(m0 + 16 * i + r) * K + h * 16;
    // B fragment (32x16 bf16, K x N): lane l holds column n0+(l&15) = W row,
    // 16 contiguous K values starting at k0 + h*16.
    #pragma unroll
    for (int j = 0; j < 4; ++j)
        wb[j] = W + (size_t)(n0 + 16 * j + r) * K + h * 16;

    for (int k0 = 0; k0 < K; k0 += 32) {
        v16bf a[4], b[4];
        #pragma unroll
        for (int i = 0; i < 4; ++i) a[i] = load_a_tile(xa[i] + k0);
        #pragma unroll
        for (int j = 0; j < 4; ++j) b[j] = *(const v16bf*)(wb[j] + k0);

        // Prefetch the streaming X rows ~1KB ahead (W stays resident in L2).
        #pragma unroll
        for (int i = 0; i < 4; ++i) __builtin_prefetch(xa[i] + k0 + 512, 0, 1);

        #pragma unroll
        for (int j = 0; j < 4; ++j)
            #pragma unroll
            for (int i = 0; i < 4; ++i)
                acc[i][j] = __builtin_amdgcn_wmma_f32_16x16x32_bf16(
                    false, a[i], false, b[j], (short)0, acc[i][j], false, false);
    }

    // Epilogue: relu, store f32, track global max (all values >= 0).
    float wmax = 0.0f;
    #pragma unroll
    for (int i = 0; i < 4; ++i) {
        #pragma unroll
        for (int j = 0; j < 4; ++j) {
            float* op = out + (size_t)(m0 + i * 16 + h * 8) * N + (n0 + j * 16 + r);
            #pragma unroll
            for (int v = 0; v < 8; ++v) {
                float val = fmaxf(acc[i][j][v], 0.0f);
                wmax = fmaxf(wmax, val);
                op[(size_t)v * N] = val;
            }
        }
    }
    #pragma unroll
    for (int o = 16; o > 0; o >>= 1) wmax = fmaxf(wmax, __shfl_xor(wmax, o, 32));
    if (lane == 0) atomicMax(maxbits, __float_as_uint(wmax));   // bits-order == float-order for >= 0
}

// --------------------------------------- power law + logprior + row softmax
__global__ void __launch_bounds__(256)
softhebb_pow_norm_kernel(float* __restrict__ a,                 // in-place [B,OUT], holds relu(a)
                         const float* __restrict__ logprior,
                         const float* __restrict__ lamb_p,
                         const unsigned* __restrict__ maxbits,
                         int ncols) {
    const size_t row = blockIdx.x;
    float* rp = a + row * (size_t)ncols;
    const float gmax = __uint_as_float(*maxbits);
    const float inv  = 1.0f / (gmax + EPS);
    const float lamb = *lamb_p;

    float sum = 0.0f;
    for (int c = threadIdx.x; c < ncols; c += blockDim.x) {
        float u  = rp[c] * inv;
        float pv = 0.0f;
        if (u > 0.0f) pv = __expf(lamb * __logf(u) + logprior[c]);
        sum += pv;
    }
    sum = block_reduce_sum256(sum);
    const float s = 1.0f / (sum + EPS);
    for (int c = threadIdx.x; c < ncols; c += blockDim.x) {
        float u  = rp[c] * inv;
        float pv = 0.0f;
        if (u > 0.0f) pv = __expf(lamb * __logf(u) + logprior[c]);
        rp[c] = pv * s;
    }
}

// ---------------------------------------------------------------- launcher
extern "C" void kernel_launch(void* const* d_in, const int* in_sizes, int n_in,
                              void* d_out, int out_size, void* d_ws, size_t ws_size,
                              hipStream_t stream) {
    const float* x  = (const float*)d_in[0];   // [B, IN]
    const float* w  = (const float*)d_in[1];   // [OUT, IN]
    const float* lp = (const float*)d_in[2];   // [OUT]
    const float* lm = (const float*)d_in[3];   // scalar lambda

    const int OUT = in_sizes[2];               // 2048
    const int IN  = in_sizes[1] / OUT;         // 2048
    const int Bn  = in_sizes[0] / IN;          // 16384

    // Workspace layout: [0,256) max cell; then x_n bf16; then W_n bf16 (~72 MB)
    char*     ws      = (char*)d_ws;
    unsigned* maxcell = (unsigned*)ws;
    __bf16*   xn      = (__bf16*)(ws + 256);
    __bf16*   wn      = (__bf16*)(ws + 256 + (size_t)Bn * IN * sizeof(__bf16));
    float*    out     = (float*)d_out;

    init_max_kernel<<<1, 1, 0, stream>>>(maxcell);
    row_l2_norm_bf16_kernel<<<Bn,  256, 0, stream>>>(x, xn, IN);
    row_l2_norm_bf16_kernel<<<OUT, 256, 0, stream>>>(w, wn, IN);

    dim3 grid(Bn / 128, OUT / 256);            // wave tile 64x64, block tile 128x256
    softhebb_wmma_gemm<<<grid, 256, 0, stream>>>(xn, wn, out, IN, OUT, maxcell);

    softhebb_pow_norm_kernel<<<Bn, 256, 0, stream>>>(out, lp, lm, maxcell, OUT);
}
